// GMN_embed_maxsim_dot_corrected_19335942766732
// MI455X (gfx1250) — matrix operand order repaired and checked by reading, not compiled, not run on hardware
//
#include <hip/hip_runtime.h>
#include <hip/hip_bf16.h>

typedef __bf16 bf16;
typedef bf16  v16bf __attribute__((ext_vector_type(16)));
typedef float v8f   __attribute__((ext_vector_type(8)));

__device__ __forceinline__ v8f zero8() { v8f z = {}; return z; }

__device__ __forceinline__ v8f wmma_bf16(v16bf a, v16bf b, v8f c) {
  // D = A(16x32 bf16) * B(32x16 bf16) + C(16x16 f32)
  return __builtin_amdgcn_wmma_f32_16x16x32_bf16(false, a, false, b, (short)0, c, false, false);
}

// A fragment (16x32 slice at k-chunk kc) from row-major LDS stage [16][ld] bf16.
// Layout (ISA 7.12.2, 16-bit A): lanes0-15 K=0..7 & 16..23, lanes16-31 K=8..15 & 24..31.
__device__ __forceinline__ v16bf ldsA(const bf16* st, int ld, int kc, int lane) {
  int m = lane & 15, h = (lane >> 4) & 1;
  const bf16* p0 = st + m * ld + kc * 32 + h * 8;
  v16bf a;
#pragma unroll
  for (int i = 0; i < 8; ++i) { a[i] = p0[i]; a[i + 8] = p0[16 + i]; }
  return a;
}

// B fragment (32x16 slice: k-chunk kc, n-tile nt) from LDS W^T [N][K] bf16.
// Layout: lanes0-15 K=0..7 & 16..23, lanes16-31 K=8..15 & 24..31; column = lane&15.
__device__ __forceinline__ v16bf ldsB(const bf16* wt, int ldk, int kc, int nt, int lane) {
  int n = lane & 15, h = (lane >> 4) & 1;
  const bf16* p = wt + (nt * 16 + n) * ldk + kc * 32 + h * 8;
  v16bf b;
#pragma unroll
  for (int i = 0; i < 8; ++i) { b[i] = p[i]; b[i + 8] = p[16 + i]; }
  return b;
}

// Load row-major global W[K][N] f32 into LDS as transposed bf16 W^T[N][K].
__device__ __forceinline__ void load_wT(bf16* dst, const float* w, int K, int Nn,
                                        int tid, int nth) {
  int tot = K * Nn;
  for (int i = tid; i < tot; i += nth) {
    int k = i / Nn, n = i % Nn;
    dst[n * K + k] = (bf16)w[i];
  }
}

// Same, but permute K blocks [0,64)<->[64,128): [ht,hf,e]@W == [hf,ht,e]@perm(W).
__device__ __forceinline__ void load_wT_swap(bf16* dst, const float* w, int K, int Nn,
                                             int tid, int nth) {
  int tot = K * Nn;
  for (int i = tid; i < tot; i += nth) {
    int k = i / Nn, n = i % Nn;
    int kp = (k < 64) ? (k + 64) : ((k < 128) ? (k - 64) : k);
    dst[n * K + kp] = (bf16)w[i];
  }
}

// -------------------- offsets (prefix sum of counts) --------------------
__global__ void k_offsets(const int* __restrict__ counts, int* __restrict__ offsets, int ng) {
  if (blockIdx.x == 0 && threadIdx.x == 0) {
    int s = 0;
    for (int i = 0; i < ng; ++i) { offsets[i] = s; s += counts[i]; }
    offsets[ng] = s;
  }
}

__global__ void k_zero(float* __restrict__ p, long long n) {
  long long i = (long long)blockIdx.x * blockDim.x + threadIdx.x;
  long long st = (long long)gridDim.x * blockDim.x;
  for (; i < n; i += st) p[i] = 0.f;
}

// -------------------- node encoder: h = nf @ W(32x64) + b --------------------
// Operates on padded row count; load rows clamped, stores unconditional.
__global__ void k_encode_nodes(const float* __restrict__ nf, const float* __restrict__ w,
                               const float* __restrict__ b, float* __restrict__ h,
                               bf16* __restrict__ h_bf, int N, int Npad) {
  extern __shared__ char sm[];
  bf16* wT = (bf16*)sm;                       // [64][32]
  bf16* stg = wT + 64 * 32;                   // per-wave 16x32
  int tid = threadIdx.x, lane = tid & 31, wave = tid >> 5, nw = blockDim.x >> 5;
  load_wT(wT, w, 32, 64, tid, blockDim.x);
  __syncthreads();
  bf16* st = stg + wave * 16 * 32;
  int tiles = Npad >> 4;
  int colb = lane & 15, rowadd = 8 * ((lane >> 4) & 1);
  for (int t = blockIdx.x * nw + wave; t < tiles; t += gridDim.x * nw) {
    for (int i = lane; i < 16 * 32; i += 32) {
      int r = i >> 5, c = i & 31, row = t * 16 + r;
      int rl = row < N ? row : N - 1;
      st[i] = (bf16)nf[rl * 32 + c];
    }
    v16bf a = ldsA(st, 32, 0, lane);
#pragma unroll
    for (int nt = 0; nt < 4; ++nt) {
      v8f acc = zero8();
      acc = wmma_bf16(a, ldsB(wT, 32, 0, nt, lane), acc);
      int col = nt * 16 + colb;
      float bias = b[col];
#pragma unroll
      for (int r = 0; r < 8; ++r) {
        int row = t * 16 + r + rowadd;
        float v = acc[r] + bias;
        h[row * 64 + col] = v;
        h_bf[row * 64 + col] = (bf16)v;
      }
    }
  }
}

// -------------------- edge encoder: e = ef @ W(16x64) + b (K padded to 32) ----
__global__ void k_encode_edges(const float* __restrict__ ef, const float* __restrict__ w,
                               const float* __restrict__ b, bf16* __restrict__ e_bf,
                               int E, int Epad) {
  extern __shared__ char sm[];
  bf16* wT = (bf16*)sm;                       // [64][32], rows k>=16 zero
  bf16* stg = wT + 64 * 32;
  int tid = threadIdx.x, lane = tid & 31, wave = tid >> 5, nw = blockDim.x >> 5;
  for (int i = tid; i < 64 * 32; i += blockDim.x) {
    int n = i / 32, k = i % 32;
    wT[n * 32 + k] = (k < 16) ? (bf16)w[k * 64 + n] : (bf16)0.f;
  }
  __syncthreads();
  bf16* st = stg + wave * 16 * 32;
  int tiles = Epad >> 4;
  int colb = lane & 15, rowadd = 8 * ((lane >> 4) & 1);
  for (int t = blockIdx.x * nw + wave; t < tiles; t += gridDim.x * nw) {
    for (int i = lane; i < 16 * 32; i += 32) {
      int r = i >> 5, c = i & 31, row = t * 16 + r;
      int rl = row < E ? row : E - 1;
      st[i] = (c < 16) ? (bf16)ef[rl * 16 + c] : (bf16)0.f;
    }
    v16bf a = ldsA(st, 32, 0, lane);
#pragma unroll
    for (int nt = 0; nt < 4; ++nt) {
      v8f acc = zero8();
      acc = wmma_bf16(a, ldsB(wT, 32, 0, nt, lane), acc);
      int col = nt * 16 + colb;
      float bias = b[col];
#pragma unroll
      for (int r = 0; r < 8; ++r) {
        int row = t * 16 + r + rowadd;
        e_bf[(size_t)row * 64 + col] = (bf16)(acc[r] + bias);
      }
    }
  }
}

// -------------------- per-edge message MLPs (fwd + rev) + segment-sum scatter --
// Gather uses the CDNA5 direct global->LDS async path (ASYNCcnt).
__global__ void k_messages(const bf16* __restrict__ h_bf, const bf16* __restrict__ e_bf,
                           const int* __restrict__ from_idx, const int* __restrict__ to_idx,
                           const float* __restrict__ w1, const float* __restrict__ b1,
                           const float* __restrict__ w2, const float* __restrict__ b2,
                           const float* __restrict__ wr1, const float* __restrict__ br1,
                           const float* __restrict__ wr2, const float* __restrict__ br2,
                           float* __restrict__ agg, int E, int Epad, int dummyRow) {
  extern __shared__ char sm[];
  bf16* w1t  = (bf16*)sm;               // [128][192]
  bf16* wr1t = w1t + 128 * 192;         // [128][192] (K blocks 0/1 swapped)
  bf16* w2t  = wr1t + 128 * 192;        // [128][128]
  bf16* wr2t = w2t + 128 * 128;         // [128][128]
  bf16* stg  = wr2t + 128 * 128;        // per-wave: A 16x192 + M 16x128
  int tid = threadIdx.x, lane = tid & 31, wave = tid >> 5, nw = blockDim.x >> 5;
  load_wT(w1t, w1, 192, 128, tid, blockDim.x);
  load_wT_swap(wr1t, wr1, 192, 128, tid, blockDim.x);
  load_wT(w2t, w2, 128, 128, tid, blockDim.x);
  load_wT(wr2t, wr2, 128, 128, tid, blockDim.x);
  __syncthreads();
  bf16* A = stg + wave * (16 * 192 + 16 * 128);
  bf16* M = A + 16 * 192;
  unsigned Abase = (unsigned)(uintptr_t)(void*)A;  // flat LDS addr, low 32b = LDS offset
  int tiles = Epad >> 4;
  int colb = lane & 15, rowadd = 8 * ((lane >> 4) & 1);
  for (int t = blockIdx.x * nw + wave; t < tiles; t += gridDim.x * nw) {
    {
      int tn = t + gridDim.x * nw;
      if (tn < tiles) __builtin_prefetch(e_bf + (size_t)tn * 16 * 64, 0, 1);
    }
    // ---- async gather [hf | ht | e] -> LDS A stage, 16B chunks, no VGPR round-trip
    // A row = 384 bytes = 24 chunks: 8 (hf) + 8 (ht) + 8 (e).
    for (int i = lane; i < 16 * 24; i += 32) {
      int r = i / 24, c = i % 24, cc = c & 7;
      int edge = t * 16 + r;
      int ec = edge < E ? edge : E - 1;
      const bf16* b0 = h_bf + (size_t)from_idx[ec] * 64;
      const bf16* b1p = h_bf + (size_t)to_idx[ec] * 64;
      const bf16* b2p = e_bf + (size_t)ec * 64;
      const bf16* base = (c < 8) ? b0 : ((c < 16) ? b1p : b2p);
      unsigned long long ga = (unsigned long long)(uintptr_t)(base + cc * 8);
      unsigned lds = Abase + (unsigned)i * 16u;
      asm volatile("global_load_async_to_lds_b128 %0, %1, off"
                   :: "v"(lds), "v"(ga) : "memory");
    }
    asm volatile("s_wait_asynccnt 0x0" ::: "memory");

    // hoisted scatter targets (tail edges -> dummy accumulator row)
    int tiTo[8], tiFrom[8];
#pragma unroll
    for (int r = 0; r < 8; ++r) {
      int edge = t * 16 + r + rowadd;
      int ec = edge < E ? edge : E - 1;
      int tt = to_idx[ec], ff = from_idx[ec];
      bool ok = edge < E;
      tiTo[r] = ok ? tt : dummyRow;
      tiFrom[r] = ok ? ff : dummyRow;
    }

    v8f acc[8];
    // ================= forward direction =================
#pragma unroll
    for (int nt = 0; nt < 8; ++nt) acc[nt] = zero8();
#pragma unroll
    for (int kc = 0; kc < 6; ++kc) {
      v16bf a = ldsA(A, 192, kc, lane);
#pragma unroll
      for (int nt = 0; nt < 8; ++nt) acc[nt] = wmma_bf16(a, ldsB(w1t, 192, kc, nt, lane), acc[nt]);
    }
#pragma unroll
    for (int nt = 0; nt < 8; ++nt) {
      int col = nt * 16 + colb;
      float bias = b1[col];
#pragma unroll
      for (int r = 0; r < 8; ++r) {
        float v = acc[nt][r] + bias;
        M[(r + rowadd) * 128 + col] = (bf16)(v > 0.f ? v : 0.f);
      }
    }
#pragma unroll
    for (int nt = 0; nt < 8; ++nt) acc[nt] = zero8();
#pragma unroll
    for (int kc = 0; kc < 4; ++kc) {
      v16bf a = ldsA(M, 128, kc, lane);
#pragma unroll
      for (int nt = 0; nt < 8; ++nt) acc[nt] = wmma_bf16(a, ldsB(w2t, 128, kc, nt, lane), acc[nt]);
    }
#pragma unroll
    for (int nt = 0; nt < 8; ++nt) {
      int col = nt * 16 + colb;
      float bias = b2[col];
#pragma unroll
      for (int r = 0; r < 8; ++r)
        atomicAdd(&agg[(size_t)tiTo[r] * 128 + col], acc[nt][r] + bias);
    }
    // ================= reverse direction (reuses A stage) =================
#pragma unroll
    for (int nt = 0; nt < 8; ++nt) acc[nt] = zero8();
#pragma unroll
    for (int kc = 0; kc < 6; ++kc) {
      v16bf a = ldsA(A, 192, kc, lane);
#pragma unroll
      for (int nt = 0; nt < 8; ++nt) acc[nt] = wmma_bf16(a, ldsB(wr1t, 192, kc, nt, lane), acc[nt]);
    }
#pragma unroll
    for (int nt = 0; nt < 8; ++nt) {
      int col = nt * 16 + colb;
      float bias = br1[col];
#pragma unroll
      for (int r = 0; r < 8; ++r) {
        float v = acc[nt][r] + bias;
        M[(r + rowadd) * 128 + col] = (bf16)(v > 0.f ? v : 0.f);
      }
    }
#pragma unroll
    for (int nt = 0; nt < 8; ++nt) acc[nt] = zero8();
#pragma unroll
    for (int kc = 0; kc < 4; ++kc) {
      v16bf a = ldsA(M, 128, kc, lane);
#pragma unroll
      for (int nt = 0; nt < 8; ++nt) acc[nt] = wmma_bf16(a, ldsB(wr2t, 128, kc, nt, lane), acc[nt]);
    }
#pragma unroll
    for (int nt = 0; nt < 8; ++nt) {
      int col = nt * 16 + colb;
      float bias = br2[col];
#pragma unroll
      for (int r = 0; r < 8; ++r)
        atomicAdd(&agg[(size_t)tiFrom[r] * 128 + col], acc[nt][r] + bias);
    }
  }
}

// -------------------- node update: h += [agg | h] @ W(192x64) + b --------------
__global__ void k_node_update(const float* __restrict__ agg, float* __restrict__ h,
                              bf16* __restrict__ h_bf, const float* __restrict__ w,
                              const float* __restrict__ b, int Npad) {
  extern __shared__ char sm[];
  bf16* wT = (bf16*)sm;                      // [64][192]
  bf16* stg = wT + 64 * 192;
  int tid = threadIdx.x, lane = tid & 31, wave = tid >> 5, nw = blockDim.x >> 5;
  load_wT(wT, w, 192, 64, tid, blockDim.x);
  __syncthreads();
  bf16* A = stg + wave * 16 * 192;
  int tiles = Npad >> 4;
  int colb = lane & 15, rowadd = 8 * ((lane >> 4) & 1);
  for (int t = blockIdx.x * nw + wave; t < tiles; t += gridDim.x * nw) {
    for (int i = lane; i < 16 * 192; i += 32) {
      int r = i / 192, c = i % 192, row = t * 16 + r;
      float v = (c < 128) ? agg[(size_t)row * 128 + c] : h[row * 64 + (c - 128)];
      A[i] = (bf16)v;
    }
    v8f acc[4];
#pragma unroll
    for (int nt = 0; nt < 4; ++nt) acc[nt] = zero8();
#pragma unroll
    for (int kc = 0; kc < 6; ++kc) {
      v16bf a = ldsA(A, 192, kc, lane);
#pragma unroll
      for (int nt = 0; nt < 4; ++nt) acc[nt] = wmma_bf16(a, ldsB(wT, 192, kc, nt, lane), acc[nt]);
    }
#pragma unroll
    for (int nt = 0; nt < 4; ++nt) {
      int col = nt * 16 + colb;
      float bias = b[col];
#pragma unroll
      for (int r = 0; r < 8; ++r) {
        int row = t * 16 + r + rowadd;
        float v = h[row * 64 + col] + acc[nt][r] + bias;
        h[row * 64 + col] = v;
        h_bf[row * 64 + col] = (bf16)v;
      }
    }
  }
}

// -------------------- gate: z = h @ W(64x128)+b ; out = z[:,64:]*sigmoid(z[:,:64])
__global__ void k_gate(const float* __restrict__ h, const float* __restrict__ w,
                       const float* __restrict__ b, bf16* __restrict__ z_bf, int Npad) {
  extern __shared__ char sm[];
  bf16* wT = (bf16*)sm;                      // [128][64]
  bf16* stg = wT + 128 * 64;
  int tid = threadIdx.x, lane = tid & 31, wave = tid >> 5, nw = blockDim.x >> 5;
  load_wT(wT, w, 64, 128, tid, blockDim.x);
  __syncthreads();
  bf16* A = stg + wave * 16 * 64;
  int tiles = Npad >> 4;
  int colb = lane & 15, rowadd = 8 * ((lane >> 4) & 1);
  for (int t = blockIdx.x * nw + wave; t < tiles; t += gridDim.x * nw) {
    for (int i = lane; i < 16 * 64; i += 32) {
      int r = i >> 6, c = i & 63, row = t * 16 + r;
      A[i] = (bf16)h[row * 64 + c];
    }
    v8f acc[8];
#pragma unroll
    for (int nt = 0; nt < 8; ++nt) acc[nt] = zero8();
#pragma unroll
    for (int kc = 0; kc < 2; ++kc) {
      v16bf a = ldsA(A, 64, kc, lane);
#pragma unroll
      for (int nt = 0; nt < 8; ++nt) acc[nt] = wmma_bf16(a, ldsB(wT, 64, kc, nt, lane), acc[nt]);
    }
#pragma unroll
    for (int nt = 0; nt < 4; ++nt) {
      int colg = nt * 16 + colb;   // gate column (z[:, :64])
      int colv = 64 + colg;        // value column (z[:, 64:])
      float bg = b[colg], bv = b[colv];
#pragma unroll
      for (int r = 0; r < 8; ++r) {
        int row = t * 16 + r + rowadd;
        float g = acc[nt][r] + bg;
        float v = acc[nt + 4][r] + bv;
        float s = 1.f / (1.f + __expf(-g));
        z_bf[row * 64 + colg] = (bf16)(v * s);
      }
    }
  }
}

// -------------------- pairwise max-sim: sum_q max_{c<nc} <z_q, z_c> ------------
// z_bf is over-allocated by 512 rows so all fragment loads are unconditional;
// invalid rows/cols are excluded by the nq/nc masks on accumulation only.
__global__ void k_scores(const bf16* __restrict__ z_bf, const int* __restrict__ counts,
                         const int* __restrict__ offsets, float* __restrict__ out) {
  __shared__ float ssum;
  int pair = blockIdx.x;
  int qg = 2 * pair, cg = 2 * pair + 1;
  int nq = counts[qg], nc = counts[cg];
  int qoff = offsets[qg], coff = offsets[cg];
  int tid = threadIdx.x, lane = tid & 31, wave = tid >> 5, nw = blockDim.x >> 5;
  if (tid == 0) ssum = 0.f;
  __syncthreads();
  float psum = 0.f;
  int m = lane & 15, hh = (lane >> 4) & 1;
  for (int qt = wave; qt < 32; qt += nw) {       // 512/16 q-tiles
    v16bf a0, a1;
    {
      int qrow = qt * 16 + m;
      const bf16* base = z_bf + ((size_t)(qoff + qrow)) * 64;
#pragma unroll
      for (int i = 0; i < 8; ++i) {
        a0[i]     = base[hh * 8 + i];
        a0[i + 8] = base[16 + hh * 8 + i];
        a1[i]     = base[32 + hh * 8 + i];
        a1[i + 8] = base[48 + hh * 8 + i];
      }
    }
    float rmax[8];
#pragma unroll
    for (int r = 0; r < 8; ++r) rmax[r] = -3.0e38f;
    int ctiles = (nc + 15) / 16;
    for (int ct = 0; ct < ctiles; ++ct) {
      int crow = ct * 16 + m;
      const bf16* cb = z_bf + ((size_t)(coff + crow)) * 64;
      v16bf b0, b1;
#pragma unroll
      for (int i = 0; i < 8; ++i) {
        b0[i]     = cb[hh * 8 + i];
        b0[i + 8] = cb[16 + hh * 8 + i];
        b1[i]     = cb[32 + hh * 8 + i];
        b1[i + 8] = cb[48 + hh * 8 + i];
      }
      v8f acc = zero8();
      acc = wmma_bf16(a0, b0, acc);
      acc = wmma_bf16(a1, b1, acc);
      int col = ct * 16 + m;       // C layout: column = lane & 15
      if (col < nc) {
#pragma unroll
        for (int r = 0; r < 8; ++r) rmax[r] = fmaxf(rmax[r], acc[r]);
      }
    }
    // max across the 16 lanes sharing each row (groups {0-15} and {16-31})
#pragma unroll
    for (int r = 0; r < 8; ++r) {
      float v = rmax[r];
      for (int msk = 8; msk >= 1; msk >>= 1) v = fmaxf(v, __shfl_xor(v, msk, 16));
      int row = qt * 16 + r + 8 * hh;
      if (m == 0 && row < nq) psum += v;
    }
  }
  for (int msk = 16; msk >= 1; msk >>= 1) psum += __shfl_xor(psum, msk, 32);
  if (lane == 0) atomicAdd(&ssum, psum);
  __syncthreads();
  if (tid == 0) out[pair] = ssum;
}

extern "C" void kernel_launch(void* const* d_in, const int* in_sizes, int n_in,
                              void* d_out, int out_size, void* d_ws, size_t ws_size,
                              hipStream_t stream) {
  const float* node_features = (const float*)d_in[0];
  const float* edge_features = (const float*)d_in[1];
  const int* from_idx   = (const int*)d_in[2];
  const int* to_idx     = (const int*)d_in[3];
  const int* counts     = (const int*)d_in[6];
  const float* w_node_enc = (const float*)d_in[7];
  const float* b_node_enc = (const float*)d_in[8];
  const float* w_edge_enc = (const float*)d_in[9];
  const float* b_edge_enc = (const float*)d_in[10];
  const float* w_msg1  = (const float*)d_in[11];
  const float* b_msg1  = (const float*)d_in[12];
  const float* w_msg2  = (const float*)d_in[13];
  const float* b_msg2  = (const float*)d_in[14];
  const float* w_rmsg1 = (const float*)d_in[15];
  const float* b_rmsg1 = (const float*)d_in[16];
  const float* w_rmsg2 = (const float*)d_in[17];
  const float* b_rmsg2 = (const float*)d_in[18];
  const float* w_node_up = (const float*)d_in[19];
  const float* b_node_up = (const float*)d_in[20];
  const float* w_agg   = (const float*)d_in[21];
  const float* b_agg   = (const float*)d_in[22];

  int N = in_sizes[0] / 32;
  int E = in_sizes[1] / 16;
  int ng = in_sizes[6];
  int Npad = (N + 15) & ~15;
  int Epad = (E + 15) & ~15;

  char* ws = (char*)d_ws;
  size_t off = 0;
  auto alloc = [&](size_t bytes) -> char* {
    char* p = ws + off;
    off = (off + bytes + 255) & ~(size_t)255;
    return p;
  };
  int*   offsets = (int*)alloc(((size_t)ng + 1) * sizeof(int));
  float* h       = (float*)alloc((size_t)Npad * 64 * sizeof(float));
  bf16*  h_bf    = (bf16*)alloc((size_t)Npad * 64 * sizeof(bf16));
  bf16*  e_bf    = (bf16*)alloc((size_t)Epad * 64 * sizeof(bf16));
  float* agg     = (float*)alloc((size_t)(Npad + 16) * 128 * sizeof(float)); // +dummy rows
  bf16*  z_bf    = (bf16*)alloc((size_t)(Npad + 512) * 64 * sizeof(bf16));   // +score slack
  (void)ws_size; (void)n_in; (void)out_size;

  k_offsets<<<1, 32, 0, stream>>>(counts, offsets, ng);

  size_t sm_enc = (size_t)64 * 32 * 2 + 4 * 16 * 32 * 2;
  k_encode_nodes<<<512, 128, sm_enc, stream>>>(node_features, w_node_enc, b_node_enc,
                                               h, h_bf, N, Npad);
  k_encode_edges<<<1024, 128, sm_enc, stream>>>(edge_features, w_edge_enc, b_edge_enc,
                                                e_bf, E, Epad);

  size_t sm_msg = (size_t)(128 * 192 * 2 + 128 * 128 * 2) * 2   // w1t+wr1t, w2t+wr2t (bf16)
                + (size_t)4 * (16 * 192 + 16 * 128) * 2;        // per-wave A + M stages
  size_t sm_up  = (size_t)64 * 192 * 2 + 4 * 16 * 192 * 2;
  for (int it = 0; it < 5; ++it) {
    k_zero<<<1024, 256, 0, stream>>>(agg, (long long)(Npad + 16) * 128);
    k_messages<<<1024, 128, sm_msg, stream>>>(h_bf, e_bf, from_idx, to_idx,
                                              w_msg1, b_msg1, w_msg2, b_msg2,
                                              w_rmsg1, b_rmsg1, w_rmsg2, b_rmsg2,
                                              agg, E, Epad, Npad);
    k_node_update<<<512, 128, sm_up, stream>>>(agg, h, h_bf, w_node_up, b_node_up, Npad);
  }

  size_t sm_gate = (size_t)128 * 64 * 2 + 4 * 16 * 64 * 2;
  k_gate<<<512, 128, sm_gate, stream>>>(h, w_agg, b_agg, z_bf, Npad);

  k_scores<<<ng / 2, 256, 0, stream>>>(z_bf, counts, offsets, (float*)d_out);
}
